// CaMoE_System_70617852281187
// MI455X (gfx1250) — matrix operand order repaired and checked by reading, not compiled, not run on hardware
//
#include <hip/hip_runtime.h>
#include <hip/hip_bf16.h>
#include <cstdint>
#include <cstddef>

// ---------------------------------------------------------------------------
// CDNA5 / gfx1250 CaMoE forward pass.
//  * weights converted f32->bf16 once per launch; activations bf16 end-to-end
//    (residual x kept f32 and accumulated in place by fused epilogues)
//  * GEMM: v_wmma_f32_16x16x32_bf16, 128x128x32 tiles, 8 wave32/block
//  * A tiles staged by the Tensor Data Mover (tensor_load_to_lds), double-
//    buffered LDS, overlapped with WMMA; B tiles staged via register-transpose
// ---------------------------------------------------------------------------

typedef __attribute__((ext_vector_type(16))) __bf16 v16bf;
typedef __attribute__((ext_vector_type(8)))  __bf16 v8bf;
typedef __attribute__((ext_vector_type(4)))  __bf16 v4bf;
typedef __attribute__((ext_vector_type(8)))  float  v8f;
typedef unsigned int u32x4 __attribute__((ext_vector_type(4)));
typedef int          i32x8 __attribute__((ext_vector_type(8)));
typedef int          i32x4 __attribute__((ext_vector_type(4)));

__device__ __forceinline__ float sigm(float x) { return 1.f / (1.f + expf(-x)); }
__device__ __forceinline__ float gelu_tanh(float x) {
    float x3 = x * x * x;
    return 0.5f * x * (1.f + tanhf(0.79788456080286535588f * (x + 0.044715f * x3)));
}

#define EPI_STORE    0
#define EPI_ADD      1
#define EPI_RELUSQ   2
#define EPI_GELU     3
#define EPI_MASKADD  4

#define BM 128
#define BN 128
#define BK 32
#define LDT 40   // LDS row stride in bf16 (32 data + 8 pad) == TDM pad output

// TDM: load one [BM x BK] bf16 tile (row stride K elements) into LDS at ldsOff,
// padding each 64B row with 16B so the LDS layout has row stride LDT.
// D# packing per CDNA5 ISA 8.3/8.4 (group0: count/lds/global/type,
// group1: data_size=2B, pad_enable, pad_interval=16dw, pad_amount=4dw,
// tensor dims, tile 32x128, stride0=K).
__device__ __forceinline__ void tdm_load_tile_A(const __bf16* gsrc, unsigned ldsOff, int K)
{
    unsigned long long ga = (unsigned long long)(uintptr_t)gsrc;
    u32x4 g0;
    g0[0] = 1u;                                            // count=1
    g0[1] = ldsOff;                                        // lds_addr
    g0[2] = (unsigned)ga;                                  // global_addr[31:0]
    g0[3] = (unsigned)((ga >> 32) & 0x01FFFFFFu) | (2u << 30); // addr[56:32] | type=2
    const unsigned K32 = (unsigned)K;
    const unsigned td1 = 1u << 20;                         // tensor_dim1 (>= M)
    i32x8 g1;
    g1[0] = (int)((1u << 16) | (1u << 20) | (3u << 22) | (3u << 25));
    g1[1] = (int)((K32 & 0xFFFFu) << 16);                  // tensor_dim0[15:0]
    g1[2] = (int)((K32 >> 16) | ((td1 & 0xFFFFu) << 16));  // td0[31:16] | td1[15:0]
    g1[3] = (int)((td1 >> 16) | ((unsigned)BK << 16));     // td1[31:16] | tile_dim0=32
    g1[4] = (int)BM;                                       // tile_dim1=128, tile_dim2=0
    g1[5] = (int)K32;                                      // tensor_dim0_stride[31:0]
    g1[6] = 0;
    g1[7] = 0;
    i32x4 gz;  gz[0] = 0; gz[1] = 0; gz[2] = 0; gz[3] = 0;
    i32x8 gz8; gz8[0] = 0; gz8[1] = 0; gz8[2] = 0; gz8[3] = 0;
    gz8[4] = 0; gz8[5] = 0; gz8[6] = 0; gz8[7] = 0;
    // 6-arg form (clang-23 / therock-10.0 headers): g0, g1, g2, g3, g4, cpol
    __builtin_amdgcn_tensor_load_to_lds(g0, g1, gz, gz, gz8, 0);
}

// M % BM == 0 (always 4096) and K % BK == 0. RAGGED=false requires N % BN == 0.
template<int EPI, bool RAGGED, bool OUTBF>
__global__ __launch_bounds__(256)
void gemm_bf16(const __bf16* __restrict__ A,   // [M,K] bf16 row-major
               const __bf16* __restrict__ Bm,  // [K,N] bf16 row-major
               void* __restrict__ Cp,          // [M,N] bf16 or f32
               int M, int Nn, int K,
               const float* __restrict__ rowScale,
               const int*   __restrict__ rowWin,
               int expert)
{
    __shared__ __bf16 AsBuf[2][BM * LDT];  // TDM double buffer  [m][k]
    __shared__ __bf16 Bs[BN * LDT];        //                    [n][k]

    const int t    = threadIdx.x;
    const int lane = t & 31;
    const int wv   = t >> 5;
    const int wm   = wv & 3;        // 4 wave rows (32 rows each)
    const int wn   = wv >> 2;       // 2 wave cols (64 cols each)
    const int half = lane >> 4;
    const int l16  = lane & 15;

    const int rowA0 = blockIdx.y * BM;
    const int colB0 = blockIdx.x * BN;

    const unsigned ldsA0 = (unsigned)(uintptr_t)(void*)&AsBuf[0][0];
    const unsigned ldsA1 = (unsigned)(uintptr_t)(void*)&AsBuf[1][0];

    const v8f vzero = {0.f,0.f,0.f,0.f,0.f,0.f,0.f,0.f};
    v8f acc[2][4];
    #pragma unroll
    for (int i = 0; i < 2; ++i)
        #pragma unroll
        for (int j = 0; j < 4; ++j) acc[i][j] = vzero;

    // B staging decomposition: 4 rows (bk + i*8) x 4 consecutive cols
    const int bk = t >> 5;          // 0..7
    const int bn = (t & 31) << 2;   // 0..124

    int  gcl[4]; bool gok[4];
    #pragma unroll
    for (int j = 0; j < 4; ++j) {
        int gc = colB0 + bn + j;
        gok[j] = gc < Nn;
        gcl[j] = gok[j] ? gc : (Nn - 1);
    }

    v4bf   bReg[4];
    __bf16 bRegS[16];

    const int nk = K / BK;

    // prologue: TDM tile 0 + B tile 0 into registers
    if (wv == 0) tdm_load_tile_A(A + (size_t)rowA0 * K, ldsA0, K);
    {
        const int k0 = 0;
        if (RAGGED) {
            #pragma unroll
            for (int i = 0; i < 4; ++i) {
                const __bf16* bp = Bm + (size_t)(k0 + bk + i * 8) * Nn;
                #pragma unroll
                for (int j = 0; j < 4; ++j) {
                    __bf16 fv = bp[gcl[j]];
                    bRegS[i * 4 + j] = gok[j] ? fv : (__bf16)0.f;
                }
            }
        } else {
            #pragma unroll
            for (int i = 0; i < 4; ++i)
                bReg[i] = *(const v4bf*)(Bm + (size_t)(k0 + bk + i * 8) * Nn + colB0 + bn);
        }
    }

    for (int kt = 0; kt < nk; ++kt) {
        // ---- stage B registers -> LDS (transpose to [n][k]) ----
        #pragma unroll
        for (int i = 0; i < 4; ++i) {
            int kk = bk + i * 8;
            #pragma unroll
            for (int j = 0; j < 4; ++j)
                Bs[(bn + j) * LDT + kk] = RAGGED ? bRegS[i * 4 + j] : bReg[i][j];
        }
        if (wv == 0) __builtin_amdgcn_s_wait_tensorcnt((short)0);  // A tile kt landed
        __syncthreads();                                           // publish As+Bs

        // ---- prefetch tile kt+1 (TDM for A, registers for B) ----
        if (kt + 1 < nk) {
            const int k0 = (kt + 1) * BK;
            if (wv == 0)
                tdm_load_tile_A(A + (size_t)rowA0 * K + k0,
                                ((kt + 1) & 1) ? ldsA1 : ldsA0, K);
            if (RAGGED) {
                #pragma unroll
                for (int i = 0; i < 4; ++i) {
                    const __bf16* bp = Bm + (size_t)(k0 + bk + i * 8) * Nn;
                    #pragma unroll
                    for (int j = 0; j < 4; ++j) {
                        __bf16 fv = bp[gcl[j]];
                        bRegS[i * 4 + j] = gok[j] ? fv : (__bf16)0.f;
                    }
                }
            } else {
                #pragma unroll
                for (int i = 0; i < 4; ++i)
                    bReg[i] = *(const v4bf*)(Bm + (size_t)(k0 + bk + i * 8) * Nn + colB0 + bn);
            }
        }

        // ---- fragments (ISA 7.12.2 wave32 16-bit layouts) ----
        const __bf16* Ab = &AsBuf[kt & 1][0];
        v16bf afr[2];
        #pragma unroll
        for (int fm = 0; fm < 2; ++fm) {
            int rr = wm * 32 + fm * 16 + l16;
            v8bf a0 = *(const v8bf*)&Ab[rr * LDT + 8 * half];
            v8bf a1 = *(const v8bf*)&Ab[rr * LDT + 16 + 8 * half];
            #pragma unroll
            for (int q = 0; q < 8; ++q) { afr[fm][q] = a0[q]; afr[fm][8 + q] = a1[q]; }
        }
        v16bf bfr[4];
        #pragma unroll
        for (int fn = 0; fn < 4; ++fn) {
            int cc = wn * 64 + fn * 16 + l16;
            v8bf b0 = *(const v8bf*)&Bs[cc * LDT + 8 * half];
            v8bf b1 = *(const v8bf*)&Bs[cc * LDT + 16 + 8 * half];
            #pragma unroll
            for (int q = 0; q < 8; ++q) { bfr[fn][q] = b0[q]; bfr[fn][8 + q] = b1[q]; }
        }

        #pragma unroll
        for (int fm = 0; fm < 2; ++fm)
            #pragma unroll
            for (int fn = 0; fn < 4; ++fn)
                acc[fm][fn] = __builtin_amdgcn_wmma_f32_16x16x32_bf16(
                    false, afr[fm], false, bfr[fn],
                    (short)0, acc[fm][fn], false, false);
        __syncthreads();
    }

    // ---- fused epilogue; C/D layout: VGPR j -> M = j + 8*half, N = lane%16 ----
    float*  Cf = (float*)Cp;
    __bf16* Cb = (__bf16*)Cp;
    #pragma unroll
    for (int fm = 0; fm < 2; ++fm) {
        int fragRow = rowA0 + wm * 32 + fm * 16 + 8 * half;
        #pragma unroll
        for (int fn = 0; fn < 4; ++fn) {
            int col = colB0 + wn * 64 + fn * 16 + l16;
            if (RAGGED && col >= Nn) continue;
            v8f c = acc[fm][fn];
            #pragma unroll
            for (int j = 0; j < 8; ++j) {
                int grow = fragRow + j;
                size_t o = (size_t)grow * Nn + col;
                float val = c[j];
                if (OUTBF) {
                    if (EPI == EPI_STORE)       Cb[o] = (__bf16)val;
                    else if (EPI == EPI_ADD)    Cb[o] = (__bf16)((float)Cb[o] + val);
                    else if (EPI == EPI_RELUSQ) { float r = val > 0.f ? val : 0.f; Cb[o] = (__bf16)(r * r); }
                    else if (EPI == EPI_GELU)   Cb[o] = (__bf16)gelu_tanh(val);
                } else {
                    if (EPI == EPI_STORE)       Cf[o] = val;
                    else if (EPI == EPI_ADD)    Cf[o] += val;
                    else if (EPI == EPI_MASKADD) {
                        if (rowWin[grow] == expert) Cf[o] += val * rowScale[grow];
                    }
                }
            }
        }
    }
}

// ---------------------------------------------------------------------------
// Support kernels (activations bf16, residual f32)
// ---------------------------------------------------------------------------

__global__ __launch_bounds__(256)
void wcvt_k(const float* __restrict__ src, __bf16* __restrict__ dst, size_t n)
{
    size_t i = ((size_t)blockIdx.x * blockDim.x + threadIdx.x) * 4;
    if (i >= n) return;
    float4 f = *(const float4*)(src + i);
    v4bf p;
    p[0] = (__bf16)f.x; p[1] = (__bf16)f.y; p[2] = (__bf16)f.z; p[3] = (__bf16)f.w;
    *(v4bf*)(dst + i) = p;
}

// C = 1024 baked in
__global__ __launch_bounds__(256)
void embed_k(const int* __restrict__ idx, const float* __restrict__ emb,
             float* __restrict__ x, int Ntok)
{
    size_t i4 = ((size_t)blockIdx.x * blockDim.x + threadIdx.x) * 4;
    size_t tot = (size_t)Ntok << 10;
    if (i4 >= tot) return;
    int n = (int)(i4 >> 10);
    int c = (int)(i4 & 1023);
    *(float4*)(x + i4) = *(const float4*)(emb + ((size_t)idx[n] << 10) + c);
}

// f32 in -> bf16 out LayerNorm
__global__ __launch_bounds__(256)
void layernorm_k(const float* __restrict__ x, const float* __restrict__ s,
                 const float* __restrict__ b, __bf16* __restrict__ y, int C)
{
    __shared__ float red[256];
    int row = blockIdx.x, t = threadIdx.x;
    const float* xr = x + (size_t)row * C;
    float sum = 0.f, sq = 0.f;
    for (int c = t; c < C; c += 256) { float v = xr[c]; sum += v; sq += v * v; }
    red[t] = sum; __syncthreads();
    for (int off = 128; off > 0; off >>= 1) { if (t < off) red[t] += red[t + off]; __syncthreads(); }
    float mean = red[0] / C;
    __syncthreads();
    red[t] = sq; __syncthreads();
    for (int off = 128; off > 0; off >>= 1) { if (t < off) red[t] += red[t + off]; __syncthreads(); }
    float var  = red[0] / C - mean * mean;
    float rstd = rsqrtf(var + 1e-5f);
    __bf16* yr = y + (size_t)row * C;
    for (int c = t; c < C; c += 256)
        yr[c] = (__bf16)((xr[c] - mean) * rstd * s[c] + b[c]);
}

__global__ __launch_bounds__(256)
void copyb_k(const __bf16* __restrict__ a, __bf16* __restrict__ b, size_t n4)
{
    size_t i = ((size_t)blockIdx.x * blockDim.x + threadIdx.x) * 4;
    if (i < n4) *(v4bf*)(b + i) = *(const v4bf*)(a + i);
}

// v = v + (v_first - v)*sigmoid(g)   (all bf16)
__global__ __launch_bounds__(256)
void gate_k(__bf16* __restrict__ v, const __bf16* __restrict__ vf,
            const __bf16* __restrict__ g, size_t n4)
{
    size_t i = ((size_t)blockIdx.x * blockDim.x + threadIdx.x) * 4;
    if (i >= n4) return;
    v4bf vv = *(const v4bf*)(v + i);
    v4bf ff = *(const v4bf*)(vf + i);
    v4bf gg = *(const v4bf*)(g + i);
    v4bf out;
    #pragma unroll
    for (int j = 0; j < 4; ++j) {
        float vj = (float)vv[j], fj = (float)ff[j];
        out[j] = (__bf16)(vj + (fj - vj) * sigm((float)gg[j]));
    }
    *(v4bf*)(v + i) = out;
}

// att_in(r) = sigmoid(r)*k*v ; state = k*v   (bf16 in/out, r in place)
__global__ __launch_bounds__(256)
void attprep_k(__bf16* __restrict__ r, const __bf16* __restrict__ k,
               const __bf16* __restrict__ v, __bf16* __restrict__ st, size_t n4)
{
    size_t i = ((size_t)blockIdx.x * blockDim.x + threadIdx.x) * 4;
    if (i >= n4) return;
    v4bf rr = *(const v4bf*)(r + i);
    v4bf kk = *(const v4bf*)(k + i);
    v4bf vv = *(const v4bf*)(v + i);
    v4bf so, ro;
    #pragma unroll
    for (int j = 0; j < 4; ++j) {
        float kv = (float)kk[j] * (float)vv[j];
        so[j] = (__bf16)kv;
        ro[j] = (__bf16)(sigm((float)rr[j]) * kv);
    }
    *(v4bf*)(st + i) = so;
    *(v4bf*)(r + i)  = ro;
}

// per-token routing (E=3 fixed); h bf16, router weights f32
__global__ __launch_bounds__(256)
void routing_k(const __bf16* __restrict__ h, const float* __restrict__ cr0,
               const float* __restrict__ cr1, const float* __restrict__ ctv,
               const float* __restrict__ Wa, const float* __restrict__ shares,
               int* __restrict__ win, float* __restrict__ scl, int C)
{
    __shared__ float red[6][256];
    int row = blockIdx.x, t = threadIdx.x;
    const __bf16* hr = h + (size_t)row * C;
    float a[6] = {0.f, 0.f, 0.f, 0.f, 0.f, 0.f};
    for (int c = t; c < C; c += 256) {
        float hv = (float)hr[c];
        a[0] += hv * cr0[c];
        a[1] += hv * cr1[c];
        a[2] += hv * ctv[c];
        a[3] += hv * Wa[c * 3 + 0];
        a[4] += hv * Wa[c * 3 + 1];
        a[5] += hv * Wa[c * 3 + 2];
    }
    #pragma unroll
    for (int i = 0; i < 6; ++i) red[i][t] = a[i];
    __syncthreads();
    for (int off = 128; off > 0; off >>= 1) {
        if (t < off)
            #pragma unroll
            for (int i = 0; i < 6; ++i) red[i][t] += red[i][t + off];
        __syncthreads();
    }
    if (t == 0) {
        float conf[3] = { sigm(red[0][0]), sigm(red[1][0]), sigm(red[2][0]) };
        float best = -1e30f; int w = 0;
        #pragma unroll
        for (int e = 0; e < 3; ++e) {
            float bid = conf[e] * shares[e] + 0.1f * red[3 + e][0];
            if (bid > best) { best = bid; w = e; }
        }
        float wc = conf[w];
        win[row] = w;
        scl[row] = wc / (wc + 1e-6f);
    }
}

// ---------------------------------------------------------------------------
// Orchestration
// ---------------------------------------------------------------------------

extern "C" void kernel_launch(void* const* d_in, const int* in_sizes, int n_in,
                              void* d_out, int out_size, void* d_ws, size_t ws_size,
                              hipStream_t stream)
{
    (void)in_sizes; (void)n_in; (void)out_size; (void)ws_size;

    constexpr int Bb = 2, T = 2048, C = 1024, Lc = 2, Vv = 50257, F = 4096,
                  Ee = 3, NRr = 2;
    constexpr int N  = Bb * T;
    const size_t NC  = (size_t)N * C;
    const size_t CC  = (size_t)C * C;

    const int*   idx    = (const int*)  d_in[0];
    const float* emb    = (const float*)d_in[1];
    const float* ln1_s  = (const float*)d_in[2];
    const float* ln1_b  = (const float*)d_in[3];
    const float* ln2_s  = (const float*)d_in[4];
    const float* ln2_b  = (const float*)d_in[5];
    const float* Wr     = (const float*)d_in[6];
    const float* Wk     = (const float*)d_in[7];
    const float* Wv     = (const float*)d_in[8];
    const float* Wg     = (const float*)d_in[9];
    const float* Wo     = (const float*)d_in[10];
    const float* W1r    = (const float*)d_in[11];
    const float* W2r    = (const float*)d_in[12];
    const float* cr     = (const float*)d_in[13];
    const float* Ws     = (const float*)d_in[14];
    const float* Wt1    = (const float*)d_in[15];
    const float* Wt2    = (const float*)d_in[16];
    const float* ct     = (const float*)d_in[17];
    const float* Wa     = (const float*)d_in[18];
    /* d_in[19] = Wb unused */
    const float* shares = (const float*)d_in[20];
    const float* lno_s  = (const float*)d_in[21];
    const float* lno_b  = (const float*)d_in[22];
    const float* headW  = (const float*)d_in[23];

    float* logits = (float*)d_out;

    // ---- workspace carve-out ----
    char* base = (char*)d_ws;
    float*  x    = (float*)base;                       // NC f32
    __bf16* hid  = (__bf16*)(base + NC * 4);           // N*F bf16
    __bf16* act  = hid  + (size_t)N * F;
    __bf16* rbuf = act  + NC;
    __bf16* kbuf = rbuf + NC;
    __bf16* vbuf = kbuf + NC;
    __bf16* vfir = vbuf + NC;
    __bf16* gbuf = vfir + NC;
    __bf16* stbf = gbuf + NC;
    int*    win  = (int*)(stbf + NC);
    float*  scl  = (float*)(win + N);
    __bf16* wp   = (__bf16*)(scl + N);

    auto take = [&](size_t n) { __bf16* p = wp; wp += n; return p; };
    __bf16* cWr   = take((size_t)Lc * CC);
    __bf16* cWk   = take((size_t)Lc * CC);
    __bf16* cWv   = take((size_t)Lc * CC);
    __bf16* cWg   = take((size_t)Lc * CC);
    __bf16* cWo   = take((size_t)Lc * CC);
    __bf16* cW1r  = take((size_t)Lc * NRr * C * F);
    __bf16* cW2r  = take((size_t)Lc * NRr * F * C);
    __bf16* cWs   = take((size_t)Lc * CC);
    __bf16* cWt1  = take((size_t)Lc * C * F);
    __bf16* cWt2  = take((size_t)Lc * F * C);
    __bf16* cHead = take((size_t)C * Vv);

    auto cvt = [&](const float* s, __bf16* d, size_t n) {
        wcvt_k<<<(int)((n / 4 + 255) / 256), 256, 0, stream>>>(s, d, n);
    };
    cvt(Wr,   cWr,   (size_t)Lc * CC);
    cvt(Wk,   cWk,   (size_t)Lc * CC);
    cvt(Wv,   cWv,   (size_t)Lc * CC);
    cvt(Wg,   cWg,   (size_t)Lc * CC);
    cvt(Wo,   cWo,   (size_t)Lc * CC);
    cvt(W1r,  cW1r,  (size_t)Lc * NRr * C * F);
    cvt(W2r,  cW2r,  (size_t)Lc * NRr * F * C);
    cvt(Ws,   cWs,   (size_t)Lc * CC);
    cvt(Wt1,  cWt1,  (size_t)Lc * C * F);
    cvt(Wt2,  cWt2,  (size_t)Lc * F * C);
    cvt(headW,cHead, (size_t)C * Vv);

    const int eltBlocks4 = (int)((NC / 4 + 255) / 256);
    dim3 bl(256);
    auto grid = [](int M, int Nn) { return dim3((Nn + BN - 1) / BN, (M + BM - 1) / BM); };

    embed_k<<<eltBlocks4, 256, 0, stream>>>(idx, emb, x, N);

    for (int l = 0; l < Lc; ++l) {
        // --- time-mix attention ---
        layernorm_k<<<N, 256, 0, stream>>>(x, ln1_s + (size_t)l * C, ln1_b + (size_t)l * C, act, C);

        gemm_bf16<EPI_STORE, false, true><<<grid(N, C), bl, 0, stream>>>(
            act, cWr + (size_t)l * CC, rbuf, N, C, C, nullptr, nullptr, 0);
        gemm_bf16<EPI_STORE, false, true><<<grid(N, C), bl, 0, stream>>>(
            act, cWk + (size_t)l * CC, kbuf, N, C, C, nullptr, nullptr, 0);
        gemm_bf16<EPI_STORE, false, true><<<grid(N, C), bl, 0, stream>>>(
            act, cWv + (size_t)l * CC, vbuf, N, C, C, nullptr, nullptr, 0);

        if (l == 0) {
            copyb_k<<<eltBlocks4, 256, 0, stream>>>(vbuf, vfir, NC);
        } else {
            gemm_bf16<EPI_STORE, false, true><<<grid(N, C), bl, 0, stream>>>(
                act, cWg + (size_t)l * CC, gbuf, N, C, C, nullptr, nullptr, 0);
            gate_k<<<eltBlocks4, 256, 0, stream>>>(vbuf, vfir, gbuf, NC);
        }

        attprep_k<<<eltBlocks4, 256, 0, stream>>>(rbuf, kbuf, vbuf, stbf, NC);

        // x += att_in @ Wo
        gemm_bf16<EPI_ADD, false, false><<<grid(N, C), bl, 0, stream>>>(
            rbuf, cWo + (size_t)l * CC, x, N, C, C, nullptr, nullptr, 0);

        // --- market routing ---
        layernorm_k<<<N, 256, 0, stream>>>(x, ln2_s + (size_t)l * C, ln2_b + (size_t)l * C, act, C);
        routing_k<<<N, 256, 0, stream>>>(act,
                                         cr + ((size_t)l * NRr + 0) * C,
                                         cr + ((size_t)l * NRr + 1) * C,
                                         ct + (size_t)l * C,
                                         Wa + (size_t)l * C * Ee,
                                         shares + (size_t)l * Ee,
                                         win, scl, C);

        // --- relu^2 router experts (dense, masked combine) ---
        for (int e = 0; e < NRr; ++e) {
            gemm_bf16<EPI_RELUSQ, false, true><<<grid(N, F), bl, 0, stream>>>(
                act, cW1r + ((size_t)l * NRr + e) * C * F, hid, N, F, C, nullptr, nullptr, 0);
            gemm_bf16<EPI_MASKADD, false, false><<<grid(N, C), bl, 0, stream>>>(
                hid, cW2r + ((size_t)l * NRr + e) * F * C, x, N, C, F, scl, win, e);
        }

        // --- transformer expert: gelu((h + state@Ws) @ Wt1) @ Wt2 ---
        copyb_k<<<eltBlocks4, 256, 0, stream>>>(act, gbuf, NC);
        gemm_bf16<EPI_ADD, false, true><<<grid(N, C), bl, 0, stream>>>(
            stbf, cWs + (size_t)l * CC, gbuf, N, C, C, nullptr, nullptr, 0);
        gemm_bf16<EPI_GELU, false, true><<<grid(N, F), bl, 0, stream>>>(
            gbuf, cWt1 + (size_t)l * C * F, hid, N, F, C, nullptr, nullptr, 0);
        gemm_bf16<EPI_MASKADD, false, false><<<grid(N, C), bl, 0, stream>>>(
            hid, cWt2 + (size_t)l * F * C, x, N, C, F, scl, win, 2);
    }

    // final LN + head (ragged N = 50257)
    layernorm_k<<<N, 256, 0, stream>>>(x, lno_s, lno_b, act, C);
    gemm_bf16<EPI_STORE, true, false><<<grid(N, Vv), bl, 0, stream>>>(
        act, cHead, logits, N, Vv, C, nullptr, nullptr, 0);
}
